// HCA_32976758898803
// MI455X (gfx1250) — compile-verified, matrix operand order checked by reading, not run on hardware
//
#include <hip/hip_runtime.h>
#include <cstddef>
#include <cstdint>

typedef __attribute__((ext_vector_type(16))) __bf16 v16bfT;
typedef __attribute__((ext_vector_type(8)))  float  v8fT;

#define WS7   7
#define NTOK  343          // 7^3
#define NWIN  64           // 4^3 windows per batch
#define BATCH 2
#define CDIM  192
#define LTOK  21952        // 28^3
#define NH    6
#define HD    32
#define HIDD  768
#define SQZ   48
#define MROWS (BATCH * NWIN * NTOK)   // 43904, multiple of 64

// ---------------------------------------------------------------------------
// helpers
// ---------------------------------------------------------------------------
__device__ __forceinline__ v16bfT load_frag16(const __bf16* rowBase, int lane) {
    // 16-bit WMMA operand layout: lanes 0-15 -> K {0..7,16..23}; lanes 16-31 -> K {8..15,24..31}
    const int kb = (lane < 16) ? 0 : 8;
    union { v16bfT v; uint4 u[2]; } r;
    r.u[0] = *(const uint4*)(rowBase + kb);
    r.u[1] = *(const uint4*)(rowBase + kb + 16);
    return r.v;
}

__device__ __forceinline__ v16bfT zero_frag16() {
    union { v16bfT v; uint4 u[2]; } r;
    r.u[0] = make_uint4(0u, 0u, 0u, 0u);
    r.u[1] = make_uint4(0u, 0u, 0u, 0u);
    return r.v;
}

__device__ __forceinline__ float gelu_exact(float v) {
    return 0.5f * v * (1.0f + erff(v * 0.70710678118654752f));
}

// LDS byte-offset of a shared-memory pointer (for async-to-LDS inline asm)
__device__ __forceinline__ uint32_t lds_offset_of(const void* p) {
    return (uint32_t)(uintptr_t)(__attribute__((address_space(3))) const void*)p;
}

// async copy of 16 bytes global -> LDS (per-lane), tracked with ASYNCcnt
__device__ __forceinline__ void async_b128_to_lds(uint32_t lds_off, const void* gptr) {
    asm volatile("global_load_async_to_lds_b128 %0, %1, off"
                 :: "v"(lds_off), "v"((uint64_t)(uintptr_t)gptr)
                 : "memory");
}

__device__ __forceinline__ void wait_asynccnt0() {
    asm volatile("s_wait_asynccnt 0x0" ::: "memory");
}

// window-permuted row -> natural token index (b*LTOK + l)
__device__ __forceinline__ int unpermute_row(int r) {
    int b     = r / (NWIN * NTOK);
    int rem   = r % (NWIN * NTOK);
    int win   = rem / NTOK;
    int inner = rem % NTOK;
    int wh = win >> 4, ww = (win >> 2) & 3, wt = win & 3;
    int ih = inner / 49, iw = (inner / 7) % 7, it = inner % 7;
    int hh = wh * 7 + ih, wv = ww * 7 + iw, tv = wt * 7 + it;
    return b * LTOK + hh * 784 + wv * 28 + tv;
}

// ---------------------------------------------------------------------------
// f32 -> bf16 conversion (weights)
// ---------------------------------------------------------------------------
__global__ void w2bf_kernel(const float* __restrict__ src, __bf16* __restrict__ dst, int n) {
    int i = blockIdx.x * blockDim.x + threadIdx.x;
    if (i < n) dst[i] = (__bf16)src[i];
}

// ---------------------------------------------------------------------------
// LayerNorm (one wave per token), optional window permutation of output rows
// ---------------------------------------------------------------------------
template<int PERM>
__global__ __launch_bounds__(32) void ln_kernel(const float* __restrict__ x,
                                                const float* __restrict__ g,
                                                const float* __restrict__ bv,
                                                __bf16* __restrict__ out) {
    int tok  = blockIdx.x;            // 0 .. MROWS-1 (b*LTOK + l)
    int lane = threadIdx.x;
    const float* row = x + (size_t)tok * CDIM;
    float s = 0.f, sq = 0.f;
    float v[6];
    #pragma unroll
    for (int j = 0; j < 6; ++j) {
        v[j] = row[lane + 32 * j];
        s += v[j]; sq += v[j] * v[j];
    }
    #pragma unroll
    for (int off = 16; off > 0; off >>= 1) {
        s  += __shfl_xor(s,  off, 32);
        sq += __shfl_xor(sq, off, 32);
    }
    float mean = s * (1.0f / CDIM);
    float var  = sq * (1.0f / CDIM) - mean * mean;
    float rstd = rsqrtf(var + 1e-5f);

    int drow;
    if (PERM) {
        int b = tok / LTOK, l = tok % LTOK;
        int h = l / 784, w = (l / 28) % 28, t = l % 28;
        int win   = ((h / 7) * 4 + (w / 7)) * 4 + (t / 7);
        int inner = ((h % 7) * 7 + (w % 7)) * 7 + (t % 7);
        drow = (b * NWIN + win) * NTOK + inner;
    } else {
        drow = tok;
    }
    __bf16* drow_p = out + (size_t)drow * CDIM;
    #pragma unroll
    for (int j = 0; j < 6; ++j) {
        int c = lane + 32 * j;
        drow_p[c] = (__bf16)((v[j] - mean) * rstd * g[c] + bv[c]);
    }
}

// ---------------------------------------------------------------------------
// Generic bf16 WMMA GEMM: C[M,N] = A[M,K] * B[K,N] (+epilogue)
// 128 threads (4 waves), tile 64x64, BK=32. Wave w computes n-strip w*16.
// A tile staged global->LDS with GLOBAL_LOAD_ASYNC_TO_LDS_B128 (ASYNCcnt).
// EPI: 0=QKV(+bias->bf16) 1=PROJ(+bias,unperm,+resid->f32)
//      2=FC1(+bias,gelu->bf16) 3=FC2(+bias,+resid->f32)
// ---------------------------------------------------------------------------
template<int EPI>
__global__ __launch_bounds__(128) void gemm_bf16_kernel(const __bf16* __restrict__ A,
                                                        const __bf16* __restrict__ B,
                                                        const float*  __restrict__ bias,
                                                        void* __restrict__ outv,
                                                        const float* __restrict__ resid,
                                                        int M, int N, int K) {
    __shared__ __bf16 sA[64 * 32];    // [row][k]
    __shared__ __bf16 sBt[64 * 32];   // [n][k] (transposed B tile)

    const int tid  = threadIdx.x;
    const int wave = tid >> 5;
    const int lane = tid & 31;
    const int tm = blockIdx.x * 64;
    const int tn = blockIdx.y * 64;

    // per-thread staging coordinates (constant over k loop)
    const int ar = tid >> 1, ahc = (tid & 1) * 16;        // A: row / 16-elem half
    const uint32_t aLds = lds_offset_of(&sA[ar * 32 + ahc]);

    v8fT acc[4] = {{}, {}, {}, {}};

    for (int k0 = 0; k0 < K; k0 += 32) {
        __syncthreads();
        // --- A tile: 64 rows x 32 k, async DMA straight into LDS ---
        async_b128_to_lds(aLds, A + (size_t)(tm + ar) * K + k0 + ahc);
        // --- B tile transposed: 32 k x 64 n ---
        #pragma unroll
        for (int it = 0; it < 2; ++it) {
            int idx = it * 128 + tid;          // 0..255
            int k   = idx >> 3;                // 0..31
            int n8  = (idx & 7) * 8;           // 0..56
            union { uint4 u; __bf16 b[8]; } t;
            t.u = *(const uint4*)(B + (size_t)(k0 + k) * N + tn + n8);
            #pragma unroll
            for (int j = 0; j < 8; ++j) sBt[(n8 + j) * 32 + k] = t.b[j];
        }
        // prefetch next k-tile into cache while this one computes
        if (k0 + 32 < K) {
            __builtin_prefetch(A + (size_t)(tm + ar) * K + k0 + 32 + ahc, 0, 1);
            __builtin_prefetch(B + (size_t)(k0 + 32 + (tid >> 2)) * N + tn + (tid & 3) * 16, 0, 1);
        }
        wait_asynccnt0();
        __syncthreads();

        v16bfT bfrag = load_frag16(&sBt[(wave * 16 + (lane & 15)) * 32], lane);
        #pragma unroll
        for (int mi = 0; mi < 4; ++mi) {
            v16bfT afrag = load_frag16(&sA[(mi * 16 + (lane & 15)) * 32], lane);
            acc[mi] = __builtin_amdgcn_wmma_f32_16x16x32_bf16(
                false, afrag, false, bfrag, (short)0, acc[mi], false, false);
        }
    }

    // --- epilogue ---
    const int n  = tn + wave * 16 + (lane & 15);
    const float bn = bias[n];
    #pragma unroll
    for (int mi = 0; mi < 4; ++mi) {
        #pragma unroll
        for (int r = 0; r < 8; ++r) {
            int m = tm + mi * 16 + r + ((lane < 16) ? 0 : 8);
            float v = acc[mi][r] + bn;
            if (EPI == 0) {            // QKV -> bf16
                ((__bf16*)outv)[(size_t)m * N + n] = (__bf16)v;
            } else if (EPI == 1) {     // PROJ: unpermute + residual -> f32
                int nt = unpermute_row(m);
                ((float*)outv)[(size_t)nt * N + n] = v + resid[(size_t)nt * N + n];
            } else if (EPI == 2) {     // FC1: gelu -> bf16
                ((__bf16*)outv)[(size_t)m * N + n] = (__bf16)gelu_exact(v);
            } else {                   // FC2: + residual -> f32
                ((float*)outv)[(size_t)m * N + n] = v + resid[(size_t)m * N + n];
            }
        }
    }
}

// ---------------------------------------------------------------------------
// Window attention: one wave per (window, head, 16-row tile)
// grid = (22 row-tiles, 6 heads, 128 windows)
// ---------------------------------------------------------------------------
__global__ __launch_bounds__(32) void attn_kernel(const __bf16* __restrict__ qkv,
                                                  const float*  __restrict__ rpb,
                                                  __bf16* __restrict__ out) {
    __shared__ float  Ssc[16][352];
    __shared__ __bf16 Pm[16][352];
    __shared__ __bf16 Vt[32][352];   // [d][token]

    const int rt   = blockIdx.x;     // 0..21
    const int head = blockIdx.y;     // 0..5
    const int w    = blockIdx.z;     // 0..127 (b*64+win)
    const int lane = threadIdx.x;
    const float scale = 0.17677669529663687f;  // 32^-0.5

    // --- stage V transposed into LDS ---
    for (int tok = lane; tok < 352; tok += 32) {
        if (tok < NTOK) {
            const __bf16* vp = qkv + ((size_t)(w * NTOK + tok)) * (3 * CDIM) + 2 * CDIM + head * HD;
            #pragma unroll
            for (int d = 0; d < HD; ++d) Vt[d][tok] = vp[d];
        } else {
            #pragma unroll
            for (int d = 0; d < HD; ++d) Vt[d][tok] = (__bf16)0.0f;
        }
    }

    // --- Q fragment (rows rt*16..rt*16+15) ---
    v16bfT qf;
    {
        int m   = lane & 15;
        int row = rt * 16 + m;
        if (row < NTOK) {
            const __bf16* p = qkv + ((size_t)(w * NTOK + row)) * (3 * CDIM)
                              + head * HD + ((lane < 16) ? 0 : 8);
            union { v16bfT v; uint4 u[2]; } r;
            r.u[0] = *(const uint4*)p;
            r.u[1] = *(const uint4*)(p + 16);
            qf = r.v;
        } else {
            qf = zero_frag16();
        }
    }

    // hoist query-row 3D coords (loop-invariant over the 22 column tiles):
    // this lane writes score elements for rows m = r + (lane<16?0:8), r=0..7
    int qi8[8], qj8[8], qk8[8];
    #pragma unroll
    for (int r = 0; r < 8; ++r) {
        int m      = r + ((lane < 16) ? 0 : 8);
        int rowTok = rt * 16 + m;
        int rc = (rowTok < NTOK) ? rowTok : (NTOK - 1);
        qi8[r] = rc / 49; qj8[r] = (rc / 7) % 7; qk8[r] = rc % 7;
    }
    __syncthreads();

    // --- scores: S = scale*(Q K^T) + bias ---
    for (int ct = 0; ct < 22; ++ct) {
        v16bfT kf;
        {
            int nm  = lane & 15;
            int tok = ct * 16 + nm;
            if (tok < NTOK) {
                const __bf16* p = qkv + ((size_t)(w * NTOK + tok)) * (3 * CDIM)
                                  + CDIM + head * HD + ((lane < 16) ? 0 : 8);
                union { v16bfT v; uint4 u[2]; } r;
                r.u[0] = *(const uint4*)p;
                r.u[1] = *(const uint4*)(p + 16);
                kf = r.v;
            } else {
                kf = zero_frag16();
            }
        }
        v8fT sc = {};
        sc = __builtin_amdgcn_wmma_f32_16x16x32_bf16(false, qf, false, kf, (short)0, sc, false, false);

        int nn    = ct * 16 + (lane & 15);
        int colOk = (nn < NTOK);
        // column coords: one divmod set per (lane, ct)
        int ci = nn / 49, cj = (nn / 7) % 7, ck = nn % 7;
        int cbase0 = 6 - ci, cbase1 = 6 - cj, cbase2 = 6 - ck;
        #pragma unroll
        for (int r = 0; r < 8; ++r) {
            int m = r + ((lane < 16) ? 0 : 8);
            float val;
            if (colOk) {
                int idx = ((qi8[r] + cbase0) * 13 + (qj8[r] + cbase1)) * 13 + (qk8[r] + cbase2);
                val = sc[r] * scale + rpb[idx * NH + head];
            } else {
                val = -1e30f;
            }
            Ssc[m][nn] = val;
        }
    }
    __syncthreads();

    // --- softmax over 343 valid columns, P in bf16 (padded cols -> 0) ---
    for (int r = 0; r < 16; ++r) {
        float mx = -1e30f;
        for (int c = lane; c < NTOK; c += 32) mx = fmaxf(mx, Ssc[r][c]);
        #pragma unroll
        for (int off = 16; off > 0; off >>= 1) mx = fmaxf(mx, __shfl_xor(mx, off, 32));
        float sum = 0.f;
        for (int c = lane; c < NTOK; c += 32) {
            float e = __expf(Ssc[r][c] - mx);
            Ssc[r][c] = e;
            sum += e;
        }
        #pragma unroll
        for (int off = 16; off > 0; off >>= 1) sum += __shfl_xor(sum, off, 32);
        float rs = 1.0f / sum;
        for (int c = lane; c < 352; c += 32)
            Pm[r][c] = (c < NTOK) ? (__bf16)(Ssc[r][c] * rs) : (__bf16)0.0f;
    }
    __syncthreads();

    // --- out = P @ V (11 k-steps of 32, two 16-wide N halves) ---
    v8fT o[2] = {{}, {}};
    for (int ks = 0; ks < 11; ++ks) {
        v16bfT pf = load_frag16(&Pm[lane & 15][ks * 32], lane);
        #pragma unroll
        for (int hh = 0; hh < 2; ++hh) {
            v16bfT vf = load_frag16(&Vt[hh * 16 + (lane & 15)][ks * 32], lane);
            o[hh] = __builtin_amdgcn_wmma_f32_16x16x32_bf16(
                false, pf, false, vf, (short)0, o[hh], false, false);
        }
    }
    #pragma unroll
    for (int hh = 0; hh < 2; ++hh) {
        #pragma unroll
        for (int r = 0; r < 8; ++r) {
            int m   = r + ((lane < 16) ? 0 : 8);
            int tok = rt * 16 + m;
            if (tok < NTOK) {
                out[((size_t)(w * NTOK + tok)) * CDIM + head * HD + hh * 16 + (lane & 15)]
                    = (__bf16)o[hh][r];
            }
        }
    }
}

// ---------------------------------------------------------------------------
// Depthwise 3D convs (channel-last bf16, f32 accumulate)
// grid = (21952, 2), block = 256; thread covers 3 channels
// ---------------------------------------------------------------------------
__global__ __launch_bounds__(256) void dwconv5_kernel(const __bf16* __restrict__ h,
                                                      const float* __restrict__ wgt,
                                                      const float* __restrict__ bias,
                                                      __bf16* __restrict__ out) {
    int sp = blockIdx.x, b = blockIdx.y;
    int az = sp % 28, ay = (sp / 28) % 28, ax = sp / 784;
    for (int c = threadIdx.x; c < HIDD; c += 256) {
        float acc = bias[c];
        const float* wc = wgt + (size_t)c * 125;
        for (int kd = 0; kd < 5; ++kd) {
            int ix = ax + kd - 2; if (ix < 0 || ix >= 28) continue;
            for (int kh = 0; kh < 5; ++kh) {
                int iy = ay + kh - 2; if (iy < 0 || iy >= 28) continue;
                for (int kw = 0; kw < 5; ++kw) {
                    int iz = az + kw - 2; if (iz < 0 || iz >= 28) continue;
                    acc += (float)h[((size_t)b * LTOK + ix * 784 + iy * 28 + iz) * HIDD + c]
                           * wc[(kd * 5 + kh) * 5 + kw];
                }
            }
        }
        out[((size_t)b * LTOK + sp) * HIDD + c] = (__bf16)acc;
    }
}

__global__ __launch_bounds__(256) void dwconv7d3_kernel(const __bf16* __restrict__ f,
                                                        const float* __restrict__ wgt,
                                                        const float* __restrict__ bias,
                                                        __bf16* __restrict__ out) {
    int sp = blockIdx.x, b = blockIdx.y;
    int az = sp % 28, ay = (sp / 28) % 28, ax = sp / 784;
    for (int c = threadIdx.x; c < HIDD; c += 256) {
        float acc = bias[c];
        const float* wc = wgt + (size_t)c * 343;
        for (int kd = 0; kd < 7; ++kd) {
            int ix = ax + 3 * (kd - 3); if (ix < 0 || ix >= 28) continue;
            for (int kh = 0; kh < 7; ++kh) {
                int iy = ay + 3 * (kh - 3); if (iy < 0 || iy >= 28) continue;
                for (int kw = 0; kw < 7; ++kw) {
                    int iz = az + 3 * (kw - 3); if (iz < 0 || iz >= 28) continue;
                    acc += (float)f[((size_t)b * LTOK + ix * 784 + iy * 28 + iz) * HIDD + c]
                           * wc[(kd * 7 + kh) * 7 + kw];
                }
            }
        }
        out[((size_t)b * LTOK + sp) * HIDD + c] = (__bf16)acc;
    }
}

// ---------------------------------------------------------------------------
// deterministic global average pool (2 stages, no atomics)
// ---------------------------------------------------------------------------
__global__ __launch_bounds__(256) void pool1_kernel(const __bf16* __restrict__ f2,
                                                    float* __restrict__ partial) {
    int blk = blockIdx.x, b = blockIdx.y;          // 64 blocks of 343 spatials
    int spStart = blk * 343;
    for (int c = threadIdx.x; c < HIDD; c += 256) {
        float s = 0.f;
        for (int sp = 0; sp < 343; ++sp)
            s += (float)f2[((size_t)b * LTOK + spStart + sp) * HIDD + c];
        partial[((size_t)(b * 64 + blk)) * HIDD + c] = s;
    }
}

__global__ __launch_bounds__(256) void pool2_kernel(const float* __restrict__ partial,
                                                    float* __restrict__ pool) {
    int b = blockIdx.x;
    for (int c = threadIdx.x; c < HIDD; c += 256) {
        float s = 0.f;
        for (int i = 0; i < 64; ++i) s += partial[((size_t)(b * 64 + i)) * HIDD + c];
        pool[b * HIDD + c] = s * (1.0f / (float)LTOK);
    }
}

// ---------------------------------------------------------------------------
// channel attention: relu bottleneck + sigmoid gate
// ---------------------------------------------------------------------------
__global__ __launch_bounds__(256) void ca_kernel(const float* __restrict__ pool,
                                                 const float* __restrict__ w1,
                                                 const float* __restrict__ b1,
                                                 const float* __restrict__ w2,
                                                 const float* __restrict__ b2,
                                                 float* __restrict__ s) {
    __shared__ float pm[HIDD];
    __shared__ float s1[SQZ];
    int b = blockIdx.x;
    for (int c = threadIdx.x; c < HIDD; c += 256) pm[c] = pool[b * HIDD + c];
    __syncthreads();
    if (threadIdx.x < SQZ) {
        float a = b1[threadIdx.x];
        const float* wr = w1 + (size_t)threadIdx.x * HIDD;
        for (int k = 0; k < HIDD; ++k) a += pm[k] * wr[k];
        s1[threadIdx.x] = fmaxf(a, 0.f);
    }
    __syncthreads();
    for (int c = threadIdx.x; c < HIDD; c += 256) {
        float a = b2[c];
        const float* wr = w2 + (size_t)c * SQZ;
        #pragma unroll
        for (int k = 0; k < SQZ; ++k) a += s1[k] * wr[k];
        s[b * HIDD + c] = 1.0f / (1.0f + __expf(-a));
    }
}

// ---------------------------------------------------------------------------
// h <- h + f2 * s[b,c]  (in place, bf16)
// ---------------------------------------------------------------------------
__global__ __launch_bounds__(256) void gate_kernel(__bf16* __restrict__ h,
                                                   const __bf16* __restrict__ f2,
                                                   const float* __restrict__ s) {
    size_t n = (size_t)BATCH * LTOK * HIDD;
    for (size_t i = (size_t)blockIdx.x * blockDim.x + threadIdx.x;
         i < n; i += (size_t)gridDim.x * blockDim.x) {
        int c = (int)(i % HIDD);
        int b = (int)(i / ((size_t)LTOK * HIDD));
        h[i] = (__bf16)((float)h[i] + (float)f2[i] * s[b * HIDD + c]);
    }
}

// ---------------------------------------------------------------------------
// host launcher
// ---------------------------------------------------------------------------
extern "C" void kernel_launch(void* const* d_in, const int* in_sizes, int n_in,
                              void* d_out, int out_size, void* d_ws, size_t ws_size,
                              hipStream_t stream) {
    (void)in_sizes; (void)n_in; (void)out_size; (void)ws_size;
    const float* x      = (const float*)d_in[0];
    const float* gamma1 = (const float*)d_in[1];
    const float* beta1  = (const float*)d_in[2];
    const float* qkv_w  = (const float*)d_in[3];
    const float* qkv_b  = (const float*)d_in[4];
    const float* rpb    = (const float*)d_in[5];
    const float* proj_w = (const float*)d_in[6];
    const float* proj_b = (const float*)d_in[7];
    const float* gamma2 = (const float*)d_in[8];
    const float* beta2  = (const float*)d_in[9];
    const float* fc1_w  = (const float*)d_in[10];
    const float* fc1_b  = (const float*)d_in[11];
    const float* dwl_w  = (const float*)d_in[12];
    const float* dwl_b  = (const float*)d_in[13];
    const float* dwg_w  = (const float*)d_in[14];
    const float* dwg_b  = (const float*)d_in[15];
    const float* ca_w1  = (const float*)d_in[16];
    const float* ca_b1  = (const float*)d_in[17];
    const float* ca_w2  = (const float*)d_in[18];
    const float* ca_b2  = (const float*)d_in[19];
    const float* fc2_w  = (const float*)d_in[20];
    const float* fc2_b  = (const float*)d_in[21];
    float* out = (float*)d_out;

    char* ws = (char*)d_ws;
    size_t off = 0;
    auto alloc = [&](size_t bytes) -> void* {
        void* p = ws + off;
        off = (off + bytes + 255) & ~(size_t)255;
        return p;
    };
    __bf16* wq_bf   = (__bf16*)alloc((size_t)CDIM * 3 * CDIM * 2);
    __bf16* wp_bf   = (__bf16*)alloc((size_t)CDIM * CDIM * 2);
    __bf16* w1_bf   = (__bf16*)alloc((size_t)CDIM * HIDD * 2);
    __bf16* w2_bf   = (__bf16*)alloc((size_t)HIDD * CDIM * 2);
    __bf16* ln1_bf  = (__bf16*)alloc((size_t)MROWS * CDIM * 2);
    __bf16* qkv_bf  = (__bf16*)alloc((size_t)MROWS * 3 * CDIM * 2);
    __bf16* attn_bf = (__bf16*)alloc((size_t)MROWS * CDIM * 2);
    float*  xres    = (float*) alloc((size_t)MROWS * CDIM * 4);
    __bf16* ln2_bf  = (__bf16*)alloc((size_t)MROWS * CDIM * 2);
    __bf16* h_bf    = (__bf16*)alloc((size_t)MROWS * HIDD * 2);
    __bf16* f1_bf   = (__bf16*)alloc((size_t)MROWS * HIDD * 2);
    __bf16* f2_bf   = (__bf16*)alloc((size_t)MROWS * HIDD * 2);
    float*  partial = (float*) alloc((size_t)BATCH * 64 * HIDD * 4);
    float*  pool    = (float*) alloc((size_t)BATCH * HIDD * 4);
    float*  sgate   = (float*) alloc((size_t)BATCH * HIDD * 4);

    // --- weight conversion ---
    auto conv = [&](const float* src, __bf16* dst, int n) {
        w2bf_kernel<<<(n + 255) / 256, 256, 0, stream>>>(src, dst, n);
    };
    conv(qkv_w, wq_bf, CDIM * 3 * CDIM);
    conv(proj_w, wp_bf, CDIM * CDIM);
    conv(fc1_w, w1_bf, CDIM * HIDD);
    conv(fc2_w, w2_bf, HIDD * CDIM);

    // --- LN1 (window-permuted output) ---
    ln_kernel<1><<<MROWS, 32, 0, stream>>>(x, gamma1, beta1, ln1_bf);

    // --- QKV GEMM: [43904,192] x [192,576] ---
    gemm_bf16_kernel<0><<<dim3(MROWS / 64, (3 * CDIM) / 64), 128, 0, stream>>>(
        ln1_bf, wq_bf, qkv_b, (void*)qkv_bf, nullptr, MROWS, 3 * CDIM, CDIM);

    // --- window attention ---
    attn_kernel<<<dim3(22, NH, BATCH * NWIN), 32, 0, stream>>>(qkv_bf, rpb, attn_bf);

    // --- proj GEMM + unpermute + residual ---
    gemm_bf16_kernel<1><<<dim3(MROWS / 64, CDIM / 64), 128, 0, stream>>>(
        attn_bf, wp_bf, proj_b, (void*)xres, x, MROWS, CDIM, CDIM);

    // --- LN2 ---
    ln_kernel<0><<<MROWS, 32, 0, stream>>>(xres, gamma2, beta2, ln2_bf);

    // --- FC1 + GELU ---
    gemm_bf16_kernel<2><<<dim3(MROWS / 64, HIDD / 64), 128, 0, stream>>>(
        ln2_bf, w1_bf, fc1_b, (void*)h_bf, nullptr, MROWS, HIDD, CDIM);

    // --- depthwise convs ---
    dwconv5_kernel<<<dim3(LTOK, BATCH), 256, 0, stream>>>(h_bf, dwl_w, dwl_b, f1_bf);
    dwconv7d3_kernel<<<dim3(LTOK, BATCH), 256, 0, stream>>>(f1_bf, dwg_w, dwg_b, f2_bf);

    // --- channel attention ---
    pool1_kernel<<<dim3(64, BATCH), 256, 0, stream>>>(f2_bf, partial);
    pool2_kernel<<<BATCH, 256, 0, stream>>>(partial, pool);
    ca_kernel<<<BATCH, 256, 0, stream>>>(pool, ca_w1, ca_b1, ca_w2, ca_b2, sgate);
    gate_kernel<<<4096, 256, 0, stream>>>(h_bf, f2_bf, sgate);

    // --- FC2 + residual -> output ---
    gemm_bf16_kernel<3><<<dim3(MROWS / 64, CDIM / 64), 128, 0, stream>>>(
        h_bf, w2_bf, fc2_b, (void*)out, xres, MROWS, CDIM, HIDD);
}
